// MIPS_74904229642848
// MI455X (gfx1250) — matrix operand order repaired and checked by reading, not compiled, not run on hardware
//
#include <hip/hip_runtime.h>
#include <hip/hip_bf16.h>
#include <math.h>

// ---- problem constants (from reference) ----
#define DD 64
#define EE 128
#define HH 64
#define BB 16
#define TT 512
#define WW 3
#define NN (BB * TT)     // 8192
#define GG 256           // 4*H gates
#define INV_TEMP 20.0f   // 1/0.05

typedef _Float16 v16h __attribute__((ext_vector_type(16)));
typedef _Float16 h8   __attribute__((ext_vector_type(8)));
typedef float    v8f  __attribute__((ext_vector_type(8)));

// ---------------------------------------------------------------------------
// window indices: replicate the jnp fp32 arithmetic exactly
// ---------------------------------------------------------------------------
__global__ void k_window(const float* __restrict__ u, int* __restrict__ i2) {
  int g = blockIdx.x * blockDim.x + threadIdx.x;      // row = b*T + t
  if (g >= NN) return;
  int t = g & (TT - 1);
  float s = 1.0f / (float)(TT - 1);
  int center = t + (int)((float)t * s);
  int lo = max(center - WW, 0);
  int hi = min(center + WW, TT);
  float count = (float)(hi - lo);
  int idx = lo + (int)floorf(u[g] * count);
  i2[g] = min(idx, hi - 1);
}

// ---------------------------------------------------------------------------
// gate pre-activations: xg[dir][row][g] = x[row]·wih[dir][g] + bih + bhh
// grid = (N/16, 2 dirs), block = 256 (one gate column per thread, 16 rows)
// ---------------------------------------------------------------------------
__global__ void k_xg(const float* __restrict__ x, int F,
                     const float* __restrict__ wih,
                     const float* __restrict__ bih,
                     const float* __restrict__ bhh,
                     float* __restrict__ xg) {
  __shared__ float xs[16 * 128];
  const int r0  = blockIdx.x * 16;
  const int dir = blockIdx.y;
  const int tid = threadIdx.x;
  for (int i = tid; i < 16 * F; i += 256) xs[i] = x[(size_t)r0 * F + i];
  __syncthreads();
  const int g = tid;                                   // 0..255
  const float* wr = wih + (size_t)(dir * GG + g) * F;
  const float bias = bih[dir * GG + g] + bhh[dir * GG + g];
  for (int r = 0; r < 16; ++r) {
    float acc = bias;
    const float* xr = xs + r * F;
    #pragma unroll 8
    for (int f = 0; f < F; ++f) acc += xr[f] * wr[f];
    xg[(size_t)dir * NN * GG + (size_t)(r0 + r) * GG + g] = acc;
  }
}

// ---------------------------------------------------------------------------
// LSTM scan. grid = (2 dirs, 2 inputs), block = 1024 threads (32 waves).
// thread (b,u) owns cell (b,u): computes its own i,f,g,o dots against
// h_prev (LDS) and Whh staged transposed in LDS (bank-conflict free).
// dynamic LDS: 64*256 floats (Whh^T) + 16*64 floats (h_prev) = 68 KB.
// ---------------------------------------------------------------------------
__global__ void k_scan(const float* __restrict__ xg,
                       const float* __restrict__ whh,
                       float* __restrict__ hout) {
  extern __shared__ float smem[];
  float* wsh   = smem;               // [k][256]
  float* hprev = smem + HH * GG;     // [16][64]
  const int dir = blockIdx.x, inp = blockIdx.y;
  const int tid = threadIdx.x;
  const int b = tid >> 6, uu = tid & 63;

  for (int i = tid; i < HH * GG; i += 1024) {
    int j = i / HH, k = i % HH;                       // whh row-major [j][k]
    wsh[k * GG + j] = whh[(size_t)(dir * GG + j) * HH + k];
  }
  hprev[tid & 1023] = 0.0f;
  float c = 0.0f;
  __syncthreads();

  const float* xgb = xg + ((size_t)(inp * 2 + dir)) * NN * GG;
  float* hob = hout + (size_t)inp * NN * EE;

  for (int step = 0; step < TT; ++step) {
    const int t = dir ? (TT - 1 - step) : step;
    const float* xr = xgb + ((size_t)(b * TT + t)) * GG;
    float ai = xr[uu], af = xr[64 + uu], ag = xr[128 + uu], ao = xr[192 + uu];
    const float* hp = hprev + b * HH;
    #pragma unroll 8
    for (int k = 0; k < HH; ++k) {
      const float hk = hp[k];
      const float* wk = wsh + k * GG + uu;
      ai += hk * wk[0];
      af += hk * wk[64];
      ag += hk * wk[128];
      ao += hk * wk[192];
    }
    const float si = 1.0f / (1.0f + __expf(-ai));
    const float sf = 1.0f / (1.0f + __expf(-af));
    const float so = 1.0f / (1.0f + __expf(-ao));
    c = sf * c + si * tanhf(ag);
    const float h = so * tanhf(c);
    __syncthreads();
    hprev[b * HH + uu] = h;
    hob[((size_t)(b * TT + t)) * EE + dir * HH + uu] = h;
    __syncthreads();
  }
}

// ---------------------------------------------------------------------------
// per-row L2 normalize (in place, fp32) + emit f16 copy. one block per row.
// ---------------------------------------------------------------------------
__global__ void k_norm(float* __restrict__ z, _Float16* __restrict__ zh) {
  __shared__ float red[128];
  const size_t row = blockIdx.x;
  const int tid = threadIdx.x;
  float v = z[row * EE + tid];
  red[tid] = v * v;
  __syncthreads();
  for (int s = 64; s > 0; s >>= 1) {
    if (tid < s) red[tid] += red[tid + s];
    __syncthreads();
  }
  const float inv = 1.0f / fmaxf(sqrtf(red[0]), 1e-12f);
  const float nv = v * inv;
  z[row * EE + tid]  = nv;
  zh[row * EE + tid] = (_Float16)nv;
}

// gather z2s[row] = z2h[b*T + i2[row]]
__global__ void k_gather(const _Float16* __restrict__ z2h,
                         const int* __restrict__ i2,
                         _Float16* __restrict__ z2s) {
  const int g = blockIdx.x * blockDim.x + threadIdx.x;   // over N*E
  const int row = g >> 7, e = g & 127;
  const int src = (row / TT) * TT + i2[row];
  z2s[(size_t)row * EE + e] = z2h[(size_t)src * EE + e];
}

// diag[row] = dot(z1[row], z2[src]) / TEMP   (fp32 operands)
__global__ void k_diag(const float* __restrict__ z1, const float* __restrict__ z2,
                       const int* __restrict__ i2, float* __restrict__ diag) {
  const int row = blockIdx.x * blockDim.x + threadIdx.x;
  if (row >= NN) return;
  const int src = (row / TT) * TT + i2[row];
  const float* a = z1 + (size_t)row * EE;
  const float* b = z2 + (size_t)src * EE;
  float acc = 0.0f;
  #pragma unroll 8
  for (int e = 0; e < EE; ++e) acc += a[e] * b[e];
  diag[row] = acc * INV_TEMP;
}

// ---------------------------------------------------------------------------
// WMMA logsumexp kernel: lse[i] = log( sum_j exp( dot(A_i,B_j)/T ) )
// |dot| <= 1 (unit rows) -> exp fits fp32 without max subtraction.
// grid = 64 blocks x 256 threads; wave w owns rows blk*128 + w*16 .. +15.
// A/B operands use contiguous k order (k = 16h + s), the same in A and B (wave32).
// ---------------------------------------------------------------------------
__global__ void k_lse(const _Float16* __restrict__ A, const _Float16* __restrict__ B,
                      float* __restrict__ lse) {
  const int wave = threadIdx.x >> 5;
  const int lane = threadIdx.x & 31;
  const int m16  = lane & 15;
  const int half = lane >> 4;
  const int rowBase = blockIdx.x * 128 + wave * 16;

  // Preload A operand: 4 chunks of K=32. Per lane (m = lane%16, h = lane/16):
  //   halves = A[m][kc*32 + 16h .. +15] (contiguous, 32B load)
  v16h a[4];
  #pragma unroll
  for (int kc = 0; kc < 4; ++kc) {
    const _Float16* ap = A + (size_t)(rowBase + m16) * EE + kc * 32;
    a[kc] = *(const v16h*)(ap + 16 * half);
  }

  float sacc[8];
  #pragma unroll
  for (int r = 0; r < 8; ++r) sacc[r] = 0.0f;

  for (int n0 = 0; n0 < NN; n0 += 16) {
    // B operand: lane holds column n = n0 + lane%16 (= row n of B matrix),
    // halves = B[n][kc*32 + 16h .. +15] (contiguous, 32B load)
    const _Float16* bp = B + (size_t)(n0 + m16) * EE + 16 * half;
    v8f c = {0.f, 0.f, 0.f, 0.f, 0.f, 0.f, 0.f, 0.f};
    #pragma unroll
    for (int kc = 0; kc < 4; ++kc) {
      v16h bv = *(const v16h*)(bp + kc * 32);
      c = __builtin_amdgcn_wmma_f32_16x16x32_f16(
              false, a[kc], false, bv, (short)0, c, false, false);
    }
    #pragma unroll
    for (int r = 0; r < 8; ++r) sacc[r] += __expf(c[r] * INV_TEMP);
  }

  // reduce across the 16 lanes of each half-group; c[r] row = r + 8*half
  #pragma unroll
  for (int r = 0; r < 8; ++r) {
    float s = sacc[r];
    s += __shfl_xor(s, 1, 32);
    s += __shfl_xor(s, 2, 32);
    s += __shfl_xor(s, 4, 32);
    s += __shfl_xor(s, 8, 32);
    if (m16 == 0) lse[rowBase + r + 8 * half] = __logf(s);
  }
}

// final scalar: mean(lse_row - diag) + mean(lse_col - diag)
__global__ void k_loss(const float* __restrict__ lr, const float* __restrict__ lc,
                       const float* __restrict__ dg, float* __restrict__ out) {
  __shared__ float red[256];
  const int tid = threadIdx.x;
  float acc = 0.0f;
  for (int i = tid; i < NN; i += 256) acc += lr[i] + lc[i] - 2.0f * dg[i];
  red[tid] = acc;
  __syncthreads();
  for (int s = 128; s > 0; s >>= 1) {
    if (tid < s) red[tid] += red[tid + s];
    __syncthreads();
  }
  if (tid == 0) out[0] = red[0] / (float)NN;
}

// ---------------------------------------------------------------------------
extern "C" void kernel_launch(void* const* d_in, const int* in_sizes, int n_in,
                              void* d_out, int out_size, void* d_ws, size_t ws_size,
                              hipStream_t stream) {
  (void)in_sizes; (void)n_in; (void)out_size; (void)ws_size;
  const float* x1   = (const float*)d_in[0];
  const float* x2   = (const float*)d_in[1];
  const float* u    = (const float*)d_in[2];
  const float* wih0 = (const float*)d_in[3];
  const float* whh0 = (const float*)d_in[4];
  const float* bih0 = (const float*)d_in[5];
  const float* bhh0 = (const float*)d_in[6];
  const float* wih1 = (const float*)d_in[7];
  const float* whh1 = (const float*)d_in[8];
  const float* bih1 = (const float*)d_in[9];
  const float* bhh1 = (const float*)d_in[10];
  float* out = (float*)d_out;

  char* ws = (char*)d_ws;
  size_t off = 0;
  auto alloc = [&](size_t bytes) -> void* {
    off = (off + 255) & ~(size_t)255;
    void* p = ws + off;
    off += bytes;
    return p;
  };
  float*    gates = (float*)alloc((size_t)2 * 2 * NN * GG * sizeof(float)); // 33.5 MB
  float*    h0    = (float*)alloc((size_t)2 * NN * EE * sizeof(float));     // 8.4 MB
  float*    z     = (float*)alloc((size_t)2 * NN * EE * sizeof(float));     // 8.4 MB
  _Float16* zh    = (_Float16*)alloc((size_t)2 * NN * EE * sizeof(_Float16));
  _Float16* z2s   = (_Float16*)alloc((size_t)NN * EE * sizeof(_Float16));
  int*      i2    = (int*)alloc((size_t)NN * sizeof(int));
  float*    lseR  = (float*)alloc((size_t)NN * sizeof(float));
  float*    lseC  = (float*)alloc((size_t)NN * sizeof(float));
  float*    diag  = (float*)alloc((size_t)NN * sizeof(float));

  const size_t scanLds = (size_t)(HH * GG + BB * HH) * sizeof(float); // 68 KB

  k_window<<<NN / 256, 256, 0, stream>>>(u, i2);

  // layer 0
  k_xg<<<dim3(NN / 16, 2), 256, 0, stream>>>(x1, DD, wih0, bih0, bhh0, gates);
  k_xg<<<dim3(NN / 16, 2), 256, 0, stream>>>(x2, DD, wih0, bih0, bhh0,
                                             gates + (size_t)2 * NN * GG);
  k_scan<<<dim3(2, 2), 1024, scanLds, stream>>>(gates, whh0, h0);

  // layer 1 (input = h0, F = 2H = 128)
  k_xg<<<dim3(NN / 16, 2), 256, 0, stream>>>(h0, EE, wih1, bih1, bhh1, gates);
  k_xg<<<dim3(NN / 16, 2), 256, 0, stream>>>(h0 + (size_t)NN * EE, EE, wih1, bih1, bhh1,
                                             gates + (size_t)2 * NN * GG);
  k_scan<<<dim3(2, 2), 1024, scanLds, stream>>>(gates, whh1, z);

  // normalize both embeddings; emit f16
  k_norm<<<2 * NN, 128, 0, stream>>>(z, zh);

  // windowed gather of z2, diagonal logits (fp32), row/col logsumexp (WMMA)
  k_gather<<<(NN * EE) / 256, 256, 0, stream>>>(zh + (size_t)NN * EE, i2, z2s);
  k_diag<<<NN / 256, 256, 0, stream>>>(z, z + (size_t)NN * EE, i2, diag);
  k_lse<<<NN / 128, 256, 0, stream>>>(zh, z2s, lseR);   // rows:  z1 @ z2s^T
  k_lse<<<NN / 128, 256, 0, stream>>>(z2s, zh, lseC);   // cols:  z2s @ z1^T

  k_loss<<<1, 256, 0, stream>>>(lseR, lseC, diag, out);
}
